// BinarizedConv2d_28518582846035
// MI455X (gfx1250) — 8-bit matrix operand order repaired and checked by reading, compiled, not run on hardware
//
#include <hip/hip_runtime.h>
#include <stdint.h>

typedef __attribute__((ext_vector_type(8))) int v8i;

#if defined(__gfx1250__) && __has_builtin(__builtin_amdgcn_global_load_async_to_lds_b128) && \
    __has_builtin(__builtin_amdgcn_s_wait_asynccnt)
#define USE_ASYNC_LDS 1
#else
#define USE_ASYNC_LDS 0
#endif

#define NB    4
#define CIN   256
#define HH    56
#define WW    56
#define HP    58
#define COUTC 256
#define CKK   2304
#define LTOT  3136
#define MTOT  12544
#define TILE_M 128
#define TILE_N 64
#define TILE_K 64
#define NPACKDW ((CKK/64) * (COUTC/16) * 32 * 8)

#define OFF_HDR  0
#define OFF_QX   256
#define OFF_QX1  (OFF_QX  + 3444736)
#define OFF_BW   (OFF_QX1 + 3444736)
#define OFF_P0   (OFF_BW  + 589824)
#define OFF_P1   (OFF_P0  + MTOT*COUTC*4)


__global__ void k_init(unsigned* hdr) {
    if (threadIdx.x < 8) hdr[threadIdx.x] = 0u;
}

__global__ void k_max_abs_x(const float* __restrict__ x, int n, unsigned* hdr) {
    __shared__ float smax[256];
    float m = 0.0f;
    for (int i = blockIdx.x * blockDim.x + threadIdx.x; i < n; i += gridDim.x * blockDim.x)
        m = fmaxf(m, fabsf(x[i]));
    smax[threadIdx.x] = m;
    __syncthreads();
    for (int s = 128; s > 0; s >>= 1) {
        if (threadIdx.x < (unsigned)s)
            smax[threadIdx.x] = fmaxf(smax[threadIdx.x], smax[threadIdx.x + s]);
        __syncthreads();
    }
    if (threadIdx.x == 0)
        atomicMax(hdr + 0, __float_as_uint(smax[0]));
}

__global__ void k_scale_sx(unsigned* hdr) {
    ((float*)hdr)[4] = fmaxf(__uint_as_float(hdr[0]) / 7.0f, 1e-12f);
}

__global__ __launch_bounds__(256)
void k_quantize(const float* __restrict__ x, const unsigned* __restrict__ hdr,
                signed char* __restrict__ qx, signed char* __restrict__ qx1) {
    __shared__ signed char tq[64][HP + 6];
    __shared__ signed char ts[64][HP + 6];
    const int n   = blockIdx.x / HP;
    const int hp  = blockIdx.x % HP;
    const int ci0 = blockIdx.y * 64;
    const float sx = ((const float*)hdr)[4];
    const bool hvalid = (hp >= 1 && hp <= HH);
    for (int i = threadIdx.x; i < 64 * HP; i += 256) {
        int cil = i / HP;
        int wp  = i - cil * HP;
        signed char q = 0, s = 0;
        if (hvalid && wp >= 1 && wp <= WW) {
            float v  = x[((n * CIN + ci0 + cil) * HH + (hp - 1)) * WW + (wp - 1)];
            float rr = fminf(fmaxf(rintf(v / sx), -7.0f), 7.0f);
            int qi = (int)rr;
            q = (signed char)qi;
            s = (signed char)((qi >= 4) ? 1 : ((qi <= -4) ? -1 : 0));
        }
        tq[cil][wp] = q;
        ts[cil][wp] = s;
    }
    __syncthreads();
    for (int i = threadIdx.x; i < HP * 64; i += 256) {
        int wp  = i >> 6;
        int cil = i & 63;
        size_t o = ((size_t)(n * HP + hp) * HP + wp) * CIN + ci0 + cil;
        qx[o]  = tq[cil][wp];
        qx1[o] = ts[cil][wp];
    }
}

__global__ void k_packb(const float* __restrict__ w, unsigned* __restrict__ pbw) {
    int idx = blockIdx.x * blockDim.x + threadIdx.x;
    if (idx >= NPACKDW) return;
    int vv    = idx & 7;
    int lane  = (idx >> 3) & 31;
    int ntile = (idx >> 8) & 15;
    int kblk  = idx >> 12;
    int half  = lane >> 4;
    int o     = ntile * 16 + (lane & 15);
    int kbase = kblk * 64 + (vv & 3) * 4 + ((vv >> 2) & 1) * 32 + half * 16;
    unsigned pk = 0;
#pragma unroll
    for (int b = 0; b < 4; ++b) {
        int kp = kbase + b;
        int r  = kp >> 8;
        int ci = kp & 255;
        int s  = (w[o * CKK + ci * 9 + r] >= 0.0f) ? 1 : -1;
        pk |= (unsigned)(s & 0xFF) << (8 * b);
    }
    pbw[idx] = pk;
}

__global__ __launch_bounds__(256)
void k_gemm(const signed char* __restrict__ qx, const signed char* __restrict__ qx1,
            const signed char* __restrict__ pbw,
            int* __restrict__ p0, int* __restrict__ p1, unsigned* hdr) {
    __shared__ signed char lA [TILE_M * TILE_K];
    __shared__ signed char lA1[TILE_M * TILE_K];
    const int tid  = threadIdx.x;
    const int bm   = blockIdx.x * TILE_M;
    const int bn   = blockIdx.y * TILE_N;
    const int lane = tid & 31;
    const int wv   = tid >> 5;
    const int wm   = (wv & 3) * 32;
    const int wn   = (wv >> 2) * 32;
    const int nt0  = (bn + wn) >> 4;

    const int arow  = tid >> 1;
    const int ahalf = (tid & 1) * 32;
    const int gm    = bm + arow;
    const int n_img = gm / LTOT;
    const int lrem  = gm - n_img * LTOT;
    const int oh    = lrem / WW;
    const int ow    = lrem - oh * WW;
    const int base_s = (n_img * HP + oh) * HP + ow;

    v8i accq[2][2] = {};
    v8i acc1[2][2] = {};

    for (int kb = 0; kb < CKK; kb += TILE_K) {
        const int r    = kb >> 8;
        const int cb   = kb & 255;
        const int kh   = r / 3;
        const int kw   = r - kh * 3;
        const int kblk = kb >> 6;
        {
            size_t aoff = (size_t)(base_s + kh * HP + kw) * CIN + cb + ahalf;
#if USE_ASYNC_LDS
            typedef int vint4 __attribute__((vector_size(16)));
            typedef __attribute__((address_space(1))) vint4 gvec4;
            typedef __attribute__((address_space(3))) vint4 lvec4;
            gvec4* gq = (gvec4*)(qx  + aoff);
            gvec4* gs = (gvec4*)(qx1 + aoff);
            lvec4* lq = (lvec4*)&lA [arow * TILE_K + ahalf];
            lvec4* ls = (lvec4*)&lA1[arow * TILE_K + ahalf];
            __builtin_amdgcn_global_load_async_to_lds_b128(gq, lq, 0,  0);
            __builtin_amdgcn_global_load_async_to_lds_b128(gq, lq, 16, 0);
            __builtin_amdgcn_global_load_async_to_lds_b128(gs, ls, 0,  0);
            __builtin_amdgcn_global_load_async_to_lds_b128(gs, ls, 16, 0);
#else
            const int4* sq = (const int4*)(qx  + aoff);
            const int4* ss = (const int4*)(qx1 + aoff);
            int4* dq = (int4*)&lA [arow * TILE_K + ahalf];
            int4* ds = (int4*)&lA1[arow * TILE_K + ahalf];
            dq[0] = sq[0];  dq[1] = sq[1];
            ds[0] = ss[0];  ds[1] = ss[1];
#endif
            if (kb + TILE_K < CKK) {
                int rn  = (kb + TILE_K) >> 8;
                int cbn = (kb + TILE_K) & 255;
                int khn = rn / 3, kwn = rn - khn * 3;
                size_t an = (size_t)(base_s + khn * HP + kwn) * CIN + cbn + ahalf;
                __builtin_prefetch(qx  + an, 0, 1);
                __builtin_prefetch(qx1 + an, 0, 1);
            }
        }
        v8i bf[2];
#pragma unroll
        for (int nt = 0; nt < 2; ++nt) {
            const int4* src = (const int4*)(pbw +
                (((size_t)(kblk * 16 + nt0 + nt) * 32 + lane) << 5));
            int4 lo = src[0], hi = src[1];
            bf[nt][0] = lo.x; bf[nt][1] = lo.y; bf[nt][2] = lo.z; bf[nt][3] = lo.w;
            bf[nt][4] = hi.x; bf[nt][5] = hi.y; bf[nt][6] = hi.z; bf[nt][7] = hi.w;
        }
        if (kb + TILE_K < CKK)
            __builtin_prefetch(pbw + (((size_t)((kblk + 1) * 16 + nt0) * 32 + lane) << 5), 0, 1);
#if USE_ASYNC_LDS
        __builtin_amdgcn_s_wait_asynccnt(0);
#endif
        __syncthreads();

        v8i af[2], a1f[2];
#pragma unroll
        for (int mt = 0; mt < 2; ++mt) {
            int row = wm + mt * 16 + (lane & 15);
            const int* baseQ = (const int*)&lA [row * TILE_K];
            const int* baseS = (const int*)&lA1[row * TILE_K];
#pragma unroll
            for (int vv = 0; vv < 8; ++vv) {
                int k = (vv & 3) * 4 + ((vv >> 2) & 1) * 32 + (lane >> 4) * 16;
                af[mt][vv]  = baseQ[k >> 2];
                a1f[mt][vv] = baseS[k >> 2];
            }
        }
#pragma unroll
        for (int mt = 0; mt < 2; ++mt)
#pragma unroll
            for (int nt = 0; nt < 2; ++nt) {
                accq[mt][nt] = __builtin_amdgcn_wmma_i32_16x16x64_iu8(
                    true, af[mt],  true, bf[nt], accq[mt][nt], false, false);
                acc1[mt][nt] = __builtin_amdgcn_wmma_i32_16x16x64_iu8(
                    true, a1f[mt], true, bf[nt], acc1[mt][nt], false, false);
            }
        __syncthreads();
    }

    int lmax0 = 0, lmax1 = 0;
#pragma unroll
    for (int mt = 0; mt < 2; ++mt)
#pragma unroll
        for (int nt = 0; nt < 2; ++nt) {
            int col = bn + wn + nt * 16 + (lane & 15);
#pragma unroll
            for (int j = 0; j < 8; ++j) {
                int row = bm + wm + mt * 16 + j + (lane >> 4) * 8;
                int v1 = acc1[mt][nt][j];
                int v0 = accq[mt][nt][j] - 4 * v1;
                p0[row * COUTC + col] = v0;
                p1[row * COUTC + col] = v1;
                int a0 = v0 < 0 ? -v0 : v0;
                int a1 = v1 < 0 ? -v1 : v1;
                lmax0 = a0 > lmax0 ? a0 : lmax0;
                lmax1 = a1 > lmax1 ? a1 : lmax1;
            }
        }
    atomicMax((int*)(hdr + 1), lmax0);
    atomicMax((int*)(hdr + 2), lmax1);
}

__global__ void k_scale_sa(unsigned* hdr) {
    float* f = (float*)hdr;
    f[5] = fmaxf((float)((int*)hdr)[1] / 31.0f, 1e-12f);
    f[6] = fmaxf((float)((int*)hdr)[2] / 31.0f, 1e-12f);
}

__global__ void k_max_out(const int* __restrict__ p0, const int* __restrict__ p1,
                          unsigned* hdr) {
    const float* f = (const float*)hdr;
    const float sx = f[4], sa0 = f[5], sa1 = f[6];
    __shared__ float smax[256];
    float m = 0.0f;
    const int n = MTOT * COUTC;
    for (int i = blockIdx.x * blockDim.x + threadIdx.x; i < n; i += gridDim.x * blockDim.x) {
        float q0 = fminf(fmaxf(rintf((float)p0[i] / sa0), -31.0f), 31.0f);
        float q1 = fminf(fmaxf(rintf((float)p1[i] / sa1), -31.0f), 31.0f);
        float ov = (q0 * sa0 + 4.0f * q1 * sa1) * sx;
        m = fmaxf(m, fabsf(ov));
    }
    smax[threadIdx.x] = m;
    __syncthreads();
    for (int s = 128; s > 0; s >>= 1) {
        if (threadIdx.x < (unsigned)s)
            smax[threadIdx.x] = fmaxf(smax[threadIdx.x], smax[threadIdx.x + s]);
        __syncthreads();
    }
    if (threadIdx.x == 0)
        atomicMax(hdr + 3, __float_as_uint(smax[0]));
}

__global__ void k_scale_so(unsigned* hdr) {
    ((float*)hdr)[7] = fmaxf(__uint_as_float(hdr[3]) / 127.0f, 1e-12f);
}

__global__ void k_final(const int* __restrict__ p0, const int* __restrict__ p1,
                        const float* __restrict__ bias, const unsigned* __restrict__ hdr,
                        float* __restrict__ out) {
    int idx = blockIdx.x * blockDim.x + threadIdx.x;
    if (idx >= MTOT * COUTC) return;
    const float* f = (const float*)hdr;
    const float sx = f[4], sa0 = f[5], sa1 = f[6], so = f[7];
    int ow = idx % WW;  int t = idx / WW;
    int oh = t % HH;    t /= HH;
    int c  = t % COUTC; int n = t / COUTC;
    int e  = (n * LTOT + oh * WW + ow) * COUTC + c;
    float q0 = fminf(fmaxf(rintf((float)p0[e] / sa0), -31.0f), 31.0f);
    float q1 = fminf(fmaxf(rintf((float)p1[e] / sa1), -31.0f), 31.0f);
    float ov = (q0 * sa0 + 4.0f * q1 * sa1) * sx;
    float qo = fminf(fmaxf(rintf(ov / so), -127.0f), 127.0f);
    out[idx] = qo * so + bias[c];
}

extern "C" void kernel_launch(void* const* d_in, const int* in_sizes, int n_in,
                              void* d_out, int out_size, void* d_ws, size_t ws_size,
                              hipStream_t stream) {
    const float* x      = (const float*)d_in[0];
    const float* weight = (const float*)d_in[1];
    const float* bias   = (const float*)d_in[2];
    float* out = (float*)d_out;
    char*  ws  = (char*)d_ws;
    unsigned*    hdr = (unsigned*)(ws + OFF_HDR);
    signed char* qx  = (signed char*)(ws + OFF_QX);
    signed char* qx1 = (signed char*)(ws + OFF_QX1);
    signed char* bw  = (signed char*)(ws + OFF_BW);
    int* p0 = (int*)(ws + OFF_P0);
    int* p1 = (int*)(ws + OFF_P1);

    k_init<<<1, 32, 0, stream>>>(hdr);
    const int nx = NB * CIN * HH * WW;
    k_max_abs_x<<<512, 256, 0, stream>>>(x, nx, hdr);
    k_scale_sx<<<1, 1, 0, stream>>>(hdr);
    dim3 qgrid(NB * HP, CIN / 64);
    k_quantize<<<qgrid, 256, 0, stream>>>(x, hdr, qx, qx1);
    k_packb<<<(NPACKDW + 255) / 256, 256, 0, stream>>>(weight, (unsigned*)bw);
    dim3 grid(MTOT / TILE_M, COUTC / TILE_N);
    k_gemm<<<grid, 256, 0, stream>>>(qx, qx1, bw, p0, p1, hdr);
    k_scale_sa<<<1, 1, 0, stream>>>(hdr);
    k_max_out<<<512, 256, 0, stream>>>(p0, p1, hdr);
    k_scale_so<<<1, 1, 0, stream>>>(hdr);
    k_final<<<(MTOT * COUTC + 255) / 256, 256, 0, stream>>>(p0, p1, bias, hdr, out);
}
